// DeltaNetLayer_72009421685068
// MI455X (gfx1250) — matrix operand order repaired and checked by reading, not compiled, not run on hardware
//
#include <hip/hip_runtime.h>
#include <hip/hip_bf16.h>

// ---------------------------------------------------------------------------
// DeltaNet layer for MI455X (gfx1250, wave32, WMMA + TDM)
//   B=2, L=1024, D=1024, H=8, HD=128
// Pipeline:
//   1) proj_gemm   : q|k|v = act(x @ W)      bf16 WMMA 16x16x32, fp32 accum
//   2) kn_beta     : k row-normalize per head; beta = sigmoid(x @ Wb)
//   3) delta_scan  : chunked delta rule (C=64), fp32 WMMA 16x16x4,
//                    128x128 state in LDS, TDM double-buffered tile prefetch
//   4) lnorm       : LayerNorm over D -> d_out
// ---------------------------------------------------------------------------

typedef float    v8f   __attribute__((ext_vector_type(8)));
typedef float    v2f   __attribute__((ext_vector_type(2)));
typedef __bf16   v16bf __attribute__((ext_vector_type(16)));
typedef unsigned v4u   __attribute__((ext_vector_type(4)));
typedef int      v4i   __attribute__((ext_vector_type(4)));
typedef int      v8i   __attribute__((ext_vector_type(8)));

#define NTOK 2048   // B*L
#define DIM  1024
#define NH   8
#define HD   128
#define CHUNK 64
#define NCHUNK (1024 / CHUNK)

#if defined(__has_builtin)
#if __has_builtin(__builtin_amdgcn_tensor_load_to_lds) && \
    __has_builtin(__builtin_amdgcn_s_wait_tensorcnt)
#define USE_TDM 1
#endif
#endif

// ---------------------------------------------------------------------------
// fp32 WMMA 16x16x4 helper: acc += A(16xKD) * B(KDx16), generic strides so the
// same helper serves normal and transposed operands staged in LDS.
// A-frag (ISA 7.12.2, 32-bit A 16x4): lane m=l%16, K = kk + 2*(l/16) + {0,1}
// B-frag (row striped across lanes):  lane n=l%16, same K pair
// ---------------------------------------------------------------------------
__device__ __forceinline__ v8f wmma_f32_tile(
    const float* __restrict__ A, int a_ms, int a_ks,
    const float* __restrict__ B, int b_ks, int b_ns,
    int KD, v8f acc, int lane)
{
    const int mn = lane & 15;
    const int hi = lane >> 4;
    for (int kk = 0; kk < KD; kk += 4) {
        const int ka = kk + 2 * hi;
        v2f a, b;
        a.x = A[mn * a_ms + ka * a_ks];
        a.y = A[mn * a_ms + (ka + 1) * a_ks];
        b.x = B[ka * b_ks + mn * b_ns];
        b.y = B[(ka + 1) * b_ks + mn * b_ns];
        acc = __builtin_amdgcn_wmma_f32_16x16x4_f32(
            /*neg_a=*/false, a, /*neg_b=*/false, b,
            /*c_mod=*/(short)0, acc, /*reuse_a=*/false, /*reuse_b=*/false);
    }
    return acc;
}

// C/D 16x16 fp32 tile layout: VGPR r holds row (r + 8*(lane/16)), col lane%16
__device__ __forceinline__ v8f load_ctile(const float* __restrict__ C, int ldc, int lane) {
    v8f c;
    const int n = lane & 15, hi = lane >> 4;
#pragma unroll
    for (int r = 0; r < 8; ++r) c[r] = C[(r + 8 * hi) * ldc + n];
    return c;
}
__device__ __forceinline__ void store_ctile(float* __restrict__ C, int ldc, v8f c, int lane) {
    const int n = lane & 15, hi = lane >> 4;
#pragma unroll
    for (int r = 0; r < 8; ++r) C[(r + 8 * hi) * ldc + n] = c[r];
}

#ifdef USE_TDM
// ---------------------------------------------------------------------------
// TDM: async DMA of a 2D fp32 tile [64 rows x 128 cols], row stride 1024
// elements, from global into LDS. Descriptor per ISA 08_async_tensor §8.3-8.6.
// Issued by one wave; tracked with TENSORcnt. This toolchain exposes the
// 6-arg builtin: (v4u g0, v8i g1, v4i g2, v4i g3, v8i extra, i32 cpol);
// the extra group is zero-filled (unused/NULL per VIMAGE VADDR4 rules).
// ---------------------------------------------------------------------------
__device__ __forceinline__ void tdm_load_tile_64x128(unsigned lds_addr,
                                                     unsigned long long gaddr)
{
    v4u g0;
    g0[0] = 1u;                                   // count=1, user descriptor
    g0[1] = lds_addr;                             // LDS byte address
    g0[2] = (unsigned)(gaddr & 0xFFFFFFFFull);    // global_addr[31:0]
    g0[3] = ((unsigned)(gaddr >> 32) & 0x01FFFFFFu) | (2u << 30); // [56:32]|type=2

    v8i g1;
    g1[0] = (int)(2u << 16);          // wg_mask=0 (not in cluster), data_size=4B
    g1[1] = (int)(128u << 16);        // atomic_barrier=0 | tensor_dim0.lo16 (128)
    g1[2] = (int)(64u << 16);         // tensor_dim0.hi16=0 | tensor_dim1.lo16 (64)
    g1[3] = (int)(128u << 16);        // tensor_dim1.hi16=0 | tile_dim0 = 128
    g1[4] = (int)(64u);               // tile_dim1 = 64, tile_dim2 = 0 (unused)
    g1[5] = (int)(1024u);             // tensor_dim0_stride.lo32 = 1024 elems
    g1[6] = 0;                        // stride0.hi16 | stride1.lo16
    g1[7] = 0;                        // stride1.hi32

    v4i g2;                           // higher dims: extent 1, unused tiles
    g2[0] = 1; g2[1] = 1; g2[2] = 0; g2[3] = 0;   // tdim2=1, tdim3=1
    v4i g3;
    g3[0] = 0; g3[1] = (int)(1u << 16); g3[2] = 0; g3[3] = 0; // tdim4=1
    v8i gz;
    gz[0] = 0; gz[1] = 0; gz[2] = 0; gz[3] = 0;
    gz[4] = 0; gz[5] = 0; gz[6] = 0; gz[7] = 0;

    __builtin_amdgcn_tensor_load_to_lds(g0, g1, g2, g3, gz, 0);
}
#endif

// ---------------------------------------------------------------------------
// Kernel 1: Y = act(x @ W), tiled 128x128x32, bf16 WMMA 16x16x32.
// grid (16, 8, 3); z selects Wq/Wk/Wv; z==2 applies SiLU.
// ---------------------------------------------------------------------------
__global__ __launch_bounds__(256) void proj_gemm(
    const float* __restrict__ x,
    const float* __restrict__ Wq, const float* __restrict__ Wk,
    const float* __restrict__ Wv,
    float* __restrict__ q, float* __restrict__ k, float* __restrict__ v)
{
    __shared__ __bf16 As[128][36];  // x tile, [m][k]
    __shared__ __bf16 Bs[128][36];  // W tile transposed, [n][k]

    const int zb = blockIdx.z;
    const float* __restrict__ W = (zb == 0) ? Wq : (zb == 1) ? Wk : Wv;
    float* __restrict__ Y = (zb == 0) ? q : (zb == 1) ? k : v;

    const int m0 = blockIdx.x * 128;
    const int n0 = blockIdx.y * 128;
    const int tid = threadIdx.x;
    const int w = tid >> 5, lane = tid & 31;
    const int wm = w >> 1, wn = w & 1;          // wave -> 32x64 sub-tile
    const int hi = lane >> 4, ln = lane & 15;

    v8f acc[2][4] = {};

    for (int kb = 0; kb < DIM; kb += 32) {
        for (int i = tid; i < 4096; i += 256) {
            const int m = i >> 5, kk = i & 31;
            As[m][kk] = (__bf16)x[(long)(m0 + m) * DIM + kb + kk];
        }
        for (int i = tid; i < 4096; i += 256) {
            const int n = i >> 5, kk = i & 31;
            Bs[n][kk] = (__bf16)W[(long)(kb + kk) * DIM + n0 + n];
        }
        __syncthreads();

        // B-frags: lane n = l%16; slot j holds K = 16*(l/16) + j
        v16bf bf[4];
#pragma unroll
        for (int nsub = 0; nsub < 4; ++nsub) {
            const int ncol = wn * 64 + nsub * 16 + ln;
#pragma unroll
            for (int j = 0; j < 16; ++j) bf[nsub][j] = Bs[ncol][hi * 16 + j];
        }
        // A-frags: lane m = l%16; pair p holds K = 16*(l/16) + 2*p + {0,1}
#pragma unroll
        for (int msub = 0; msub < 2; ++msub) {
            const int mrow = wm * 32 + msub * 16 + ln;
            v16bf af;
#pragma unroll
            for (int p = 0; p < 8; ++p) {
                const int K = 2 * p + hi * 16;
                af[2 * p]     = As[mrow][K];
                af[2 * p + 1] = As[mrow][K + 1];
            }
#pragma unroll
            for (int nsub = 0; nsub < 4; ++nsub)
                acc[msub][nsub] = __builtin_amdgcn_wmma_f32_16x16x32_bf16(
                    false, af, false, bf[nsub], (short)0, acc[msub][nsub], false, false);
        }
        __syncthreads();
    }

#pragma unroll
    for (int msub = 0; msub < 2; ++msub)
#pragma unroll
        for (int nsub = 0; nsub < 4; ++nsub)
#pragma unroll
            for (int r = 0; r < 8; ++r) {
                float y = acc[msub][nsub][r];
                if (zb == 2) y = y / (1.f + __expf(-y));   // SiLU for v
                const int row = m0 + wm * 32 + msub * 16 + r + 8 * hi;
                const int col = n0 + wn * 64 + nsub * 16 + ln;
                Y[(long)row * DIM + col] = y;
            }
}

// ---------------------------------------------------------------------------
// Kernel 2: per-token: beta[h] = sigmoid(x . Wb[:,h]); k head-rows normalized.
// ---------------------------------------------------------------------------
__global__ __launch_bounds__(128) void kn_beta(
    const float* __restrict__ x, const float* __restrict__ Wb,
    float* __restrict__ k, float* __restrict__ beta)
{
    __shared__ float red[128];
    const int t = blockIdx.x, i = threadIdx.x;

    float accb[NH] = {};
    for (int d = i; d < DIM; d += 128) {
        const float xv = x[(long)t * DIM + d];
#pragma unroll
        for (int h = 0; h < NH; ++h) accb[h] += xv * Wb[d * NH + h];
    }
#pragma unroll
    for (int h = 0; h < NH; ++h) {
        red[i] = accb[h]; __syncthreads();
        for (int s = 64; s > 0; s >>= 1) { if (i < s) red[i] += red[i + s]; __syncthreads(); }
        if (i == 0) beta[t * NH + h] = 1.f / (1.f + __expf(-red[0]));
        __syncthreads();
    }
#pragma unroll
    for (int h = 0; h < NH; ++h) {
        const long gi = (long)t * DIM + h * HD + i;
        const float val = k[gi];
        red[i] = val * val; __syncthreads();
        for (int s = 64; s > 0; s >>= 1) { if (i < s) red[i] += red[i + s]; __syncthreads(); }
        const float nrm = sqrtf(red[0]);
        __syncthreads();
        k[gi] = val / fmaxf(nrm, 1e-12f);
    }
}

// ---------------------------------------------------------------------------
// Kernel 3: chunked delta-rule scan. One workgroup per (b,h); 8 wave32s.
// LDS: S[128x128] + double-buffered Q/K/V chunk tiles + O/A scratch (~304 KB;
// needs CDNA5's 320 KB/WGP). TDM prefetches chunk c+1 while chunk c computes.
// fp32 WMMA 16x16x4 for all matrix work (1024-step recurrence => keep fp32).
// ---------------------------------------------------------------------------
__global__ __launch_bounds__(256) void delta_scan(
    const float* __restrict__ q, const float* __restrict__ k,
    const float* __restrict__ v, const float* __restrict__ beta,
    float* __restrict__ o)
{
    extern __shared__ float sm[];
    float* S    = sm;                   // 128*128
    float* Qb0  = S + 16384;            // 64*128 each
    float* Kb0  = Qb0 + 8192;
    float* Vb0  = Kb0 + 8192;
    float* Qb1  = Vb0 + 8192;
    float* Kb1  = Qb1 + 8192;
    float* Vb1  = Kb1 + 8192;
    float* Oc   = Vb1 + 8192;           // 64*128  (Q @ S0)
    float* Am   = Oc + 8192;            // 64*64   (KK^T -> L, later QK^T masked)
    float* bet  = Am + 4096;            // 64

    const int bh = blockIdx.x;
    const int bIdx = bh >> 3, h = bh & 7;
    const int tid = threadIdx.x;
    const int w = tid >> 5, lane = tid & 31;
    const int n15 = lane & 15, hi = lane >> 4;
    const long tok0 = (long)bIdx * 1024;
    const int col0 = h * HD;

    for (int i = tid; i < 16384; i += 256) S[i] = 0.f;

#ifdef USE_TDM
    // pre-issue chunk 0 tile loads (one wave; TENSORcnt-tracked)
    if (w == 0) {
        const unsigned long long eb =
            ((unsigned long long)(tok0)*DIM + (unsigned long long)col0) * 4ull;
        tdm_load_tile_64x128((unsigned)(unsigned long long)Qb0,
                             (unsigned long long)(const void*)q + eb);
        tdm_load_tile_64x128((unsigned)(unsigned long long)Kb0,
                             (unsigned long long)(const void*)k + eb);
        tdm_load_tile_64x128((unsigned)(unsigned long long)Vb0,
                             (unsigned long long)(const void*)v + eb);
    }
#endif
    __syncthreads();

    for (int c = 0; c < NCHUNK; ++c) {
        float* Qc = (c & 1) ? Qb1 : Qb0;
        float* Kc = (c & 1) ? Kb1 : Kb0;
        float* U  = (c & 1) ? Vb1 : Vb0;

#ifdef USE_TDM
        if (w == 0) __builtin_amdgcn_s_wait_tensorcnt(0);  // chunk c in LDS
        __syncthreads();                                   // publish to all waves
        if (tid < CHUNK) bet[tid] = beta[(tok0 + c * CHUNK + tid) * NH + h];
        if (w == 0 && (c + 1) < NCHUNK) {                  // prefetch chunk c+1
            float* Qn = (c & 1) ? Qb0 : Qb1;
            float* Kn = (c & 1) ? Kb0 : Kb1;
            float* Vn = (c & 1) ? Vb0 : Vb1;
            const unsigned long long eb =
                ((unsigned long long)(tok0 + (c + 1) * CHUNK) * DIM +
                 (unsigned long long)col0) * 4ull;
            tdm_load_tile_64x128((unsigned)(unsigned long long)Qn,
                                 (unsigned long long)(const void*)q + eb);
            tdm_load_tile_64x128((unsigned)(unsigned long long)Kn,
                                 (unsigned long long)(const void*)k + eb);
            tdm_load_tile_64x128((unsigned)(unsigned long long)Vn,
                                 (unsigned long long)(const void*)v + eb);
        }
#else
        for (int i = tid; i < CHUNK * HD; i += 256) {
            const int row = i >> 7, d = i & 127;
            const long g = (tok0 + c * CHUNK + row) * DIM + col0 + d;
            Qc[i] = q[g]; Kc[i] = k[g]; U[i] = v[g];
        }
        if (tid < CHUNK) bet[tid] = beta[(tok0 + c * CHUNK + tid) * NH + h];
        __syncthreads();
#endif

        // ---- U -= K@S0 ; Oc = Q@S0  (64 tiles over 8 waves) -------------
        for (int idx = w; idx < 64; idx += 8) {
            const int which = idx >> 5;            // 0: K->U, 1: Q->Oc
            const int t = idx & 31;
            const int mt = t >> 3, nt = t & 7;
            const float* Aop = (which ? Qc : Kc) + mt * 16 * HD;
            v8f acc = {};
            acc = wmma_f32_tile(Aop, HD, 1, S + nt * 16, HD, 1, HD, acc, lane);
            float* Cp = (which ? Oc : U) + (mt * 16) * HD + nt * 16;
            if (which) {
#pragma unroll
                for (int r = 0; r < 8; ++r) Cp[(r + 8 * hi) * HD + n15] = acc[r];
            } else {
#pragma unroll
                for (int r = 0; r < 8; ++r) Cp[(r + 8 * hi) * HD + n15] -= acc[r];
            }
        }
        __syncthreads();

        // ---- Am = K K^T  (16 tiles) -------------------------------------
        for (int idx = w; idx < 16; idx += 8) {
            const int mt = idx >> 2, nt = idx & 3;
            v8f acc = {};
            acc = wmma_f32_tile(Kc + mt * 16 * HD, HD, 1,
                                Kc + nt * 16 * HD, 1, HD, HD, acc, lane);
            store_ctile(Am + (mt * 16) * 64 + nt * 16, 64, acc, lane);
        }
        __syncthreads();

        // ---- L = diag(beta)*stril(Am);  rhs' = diag(beta)*U -------------
        for (int i = tid; i < 4096; i += 256) {
            const int r = i >> 6, cc = i & 63;
            Am[i] = (cc < r) ? bet[r] * Am[i] : 0.f;
        }
        for (int i = tid; i < CHUNK * HD; i += 256) U[i] *= bet[i >> 7];
        __syncthreads();

        // ---- forward substitution: (I + L) u = rhs', 16-row blocks ------
        for (int bi = 0; bi < 4; ++bi) {
            if (bi > 0) {   // off-diagonal block GEMMs; wave w owns n-tile w
                v8f acc = {};
                for (int bj = 0; bj < bi; ++bj)
                    acc = wmma_f32_tile(Am + (bi * 16) * 64 + bj * 16, 64, 1,
                                        U + (bj * 16) * HD + w * 16, HD, 1,
                                        16, acc, lane);
#pragma unroll
                for (int r = 0; r < 8; ++r)
                    U[(bi * 16 + r + 8 * hi) * HD + w * 16 + n15] -= acc[r];
            }
            __syncthreads();
            for (int r = 1; r < 16; ++r) {          // in-block sequential
                if (tid < HD) {
                    const int rowg = bi * 16 + r;
                    float s = 0.f;
                    for (int j = 0; j < r; ++j)
                        s += Am[rowg * 64 + bi * 16 + j] * U[(bi * 16 + j) * HD + tid];
                    U[rowg * HD + tid] -= s;
                }
                __syncthreads();
            }
        }

        // ---- Am = tril(Q K^T) (incl. diagonal, unscaled) ----------------
        for (int idx = w; idx < 16; idx += 8) {
            const int mt = idx >> 2, nt = idx & 3;
            v8f acc = {};
            acc = wmma_f32_tile(Qc + mt * 16 * HD, HD, 1,
                                Kc + nt * 16 * HD, 1, HD, HD, acc, lane);
#pragma unroll
            for (int r = 0; r < 8; ++r) {
                const int row = mt * 16 + r + 8 * hi;
                const int cg = nt * 16 + n15;
                Am[row * 64 + cg] = (cg <= row) ? acc[r] : 0.f;
            }
        }
        __syncthreads();

        // ---- O = Oc + Am @ u  -> global ---------------------------------
        for (int idx = w; idx < 32; idx += 8) {
            const int mt = idx >> 3, nt = idx & 7;
            v8f acc = load_ctile(Oc + (mt * 16) * HD + nt * 16, HD, lane);
            acc = wmma_f32_tile(Am + (mt * 16) * 64, 64, 1,
                                U + nt * 16, HD, 1, CHUNK, acc, lane);
#pragma unroll
            for (int r = 0; r < 8; ++r) {
                const int row = mt * 16 + r + 8 * hi;
                const long g = (tok0 + c * CHUNK + row) * DIM + col0 + nt * 16 + n15;
                o[g] = acc[r];
            }
        }

        // ---- S += K^T @ u  (64 tiles) -----------------------------------
        for (int idx = w; idx < 64; idx += 8) {
            const int mt = idx >> 3, nt = idx & 7;
            v8f acc = load_ctile(S + (mt * 16) * HD + nt * 16, HD, lane);
            acc = wmma_f32_tile(Kc + mt * 16, 1, HD,      // K^T: A(m,k)=Kc[k][m]
                                U + nt * 16, HD, 1, CHUNK, acc, lane);
            store_ctile(S + (mt * 16) * HD + nt * 16, HD, acc, lane);
        }
        __syncthreads();
    }
}

// ---------------------------------------------------------------------------
// Kernel 4: LayerNorm over D per token.
// ---------------------------------------------------------------------------
__global__ __launch_bounds__(256) void lnorm(
    const float* __restrict__ o, const float* __restrict__ lw,
    const float* __restrict__ lb, float* __restrict__ out)
{
    __shared__ float r1[256], r2[256];
    const int t = blockIdx.x, i = threadIdx.x;
    float s = 0.f, ss = 0.f;
    for (int d = i; d < DIM; d += 256) {
        const float vv = o[(long)t * DIM + d];
        s += vv; ss += vv * vv;
    }
    r1[i] = s; r2[i] = ss; __syncthreads();
    for (int st = 128; st > 0; st >>= 1) {
        if (i < st) { r1[i] += r1[i + st]; r2[i] += r2[i + st]; }
        __syncthreads();
    }
    const float mu = r1[0] * (1.f / DIM);
    const float var = r2[0] * (1.f / DIM) - mu * mu;
    const float rstd = rsqrtf(var + 1e-5f);
    for (int d = i; d < DIM; d += 256)
        out[(long)t * DIM + d] = (o[(long)t * DIM + d] - mu) * rstd * lw[d] + lb[d];
}

// ---------------------------------------------------------------------------
extern "C" void kernel_launch(void* const* d_in, const int* in_sizes, int n_in,
                              void* d_out, int out_size, void* d_ws, size_t ws_size,
                              hipStream_t stream)
{
    const float* x  = (const float*)d_in[0];
    const float* Wq = (const float*)d_in[1];
    const float* Wk = (const float*)d_in[2];
    const float* Wv = (const float*)d_in[3];
    const float* Wb = (const float*)d_in[4];
    const float* lw = (const float*)d_in[5];
    const float* lb = (const float*)d_in[6];
    float* out = (float*)d_out;

    char* ws = (char*)d_ws;
    const size_t SZ = (size_t)NTOK * DIM * sizeof(float);   // 8 MB
    float* q    = (float*)(ws);
    float* k    = (float*)(ws + SZ);
    float* v    = (float*)(ws + 2 * SZ);
    float* o    = (float*)(ws + 3 * SZ);
    float* beta = (float*)(ws + 4 * SZ);                    // 64 KB

    proj_gemm<<<dim3(16, 8, 3), 256, 0, stream>>>(x, Wq, Wk, Wv, q, k, v);
    kn_beta<<<NTOK, 128, 0, stream>>>(x, Wb, k, beta);

    // S + 2x(Q,K,V) + Oc + Am + beta = (16384 + 6*8192 + 8192 + 4096 + 64) floats
    const size_t smem = (16384 + 6 * 8192 + 8192 + 4096 + 64) * sizeof(float); // ~304 KB
    delta_scan<<<16, 256, smem, stream>>>(q, k, v, beta, o);

    lnorm<<<NTOK, 256, 0, stream>>>(o, lw, lb, out);
}
